// VNMaxPool_8839042695332
// MI455X (gfx1250) — matrix operand order repaired and checked by reading, not compiled, not run on hardware
//
#include <hip/hip_runtime.h>
#include <hip/hip_bf16.h>
#include <stdint.h>

// Problem constants from the reference
constexpr int Bd = 4;
constexpr int Nd = 1024;
constexpr int Kd = 32;    // samples per point
constexpr int Cd = 128;   // channels

typedef __attribute__((ext_vector_type(16))) __bf16 v16bf;
typedef __attribute__((ext_vector_type(8)))  float  v8f;

union V16B { v16bf v; uint4 q[2]; };
union V8F  { v8f   v; float f[8]; };

// fp32 -> bf16 with round-to-nearest-even
static __device__ __forceinline__ unsigned int f2bf(float f) {
    unsigned int u = __float_as_uint(f);
    unsigned int r = u + 0x7FFFu + ((u >> 16) & 1u);
    return r >> 16;
}

// ---------------------------------------------------------------------------
// Prep: wt[0..16383] = W0^T (bf16), wt[16384..32767] = W1^T (bf16).
// Transposed so a WMMA B-operand lane reads two contiguous 16B runs along K.
// ---------------------------------------------------------------------------
__global__ void prep_WT(const float* __restrict__ W0,
                        const float* __restrict__ W1,
                        unsigned short* __restrict__ wt) {
    int g = blockIdx.x * blockDim.x + threadIdx.x;
    int stride = gridDim.x * blockDim.x;
    for (int i = g; i < 2 * Cd * Cd; i += stride) {
        int w   = i / (Cd * Cd);
        int e   = i % (Cd * Cd);
        int row = e / Cd;   // c' (GEMM K dim)
        int col = e % Cd;   // c  (GEMM N dim)
        float v = (w == 0 ? W0 : W1)[e];            // coalesced read
        wt[w * Cd * Cd + col * Cd + row] = (unsigned short)f2bf(v);
    }
}

// ---------------------------------------------------------------------------
// Main fused kernel: one workgroup (8 wave32s) per (b,n).
// ---------------------------------------------------------------------------
__global__ void __launch_bounds__(256)
vn_maxpool_kernel(const float* __restrict__ x0,
                  const float* __restrict__ x1,
                  const float* __restrict__ b0,
                  const unsigned short* __restrict__ wt,
                  float* __restrict__ out0,
                  float* __restrict__ out1) {
    // x in bf16, layout [d][k][c] packed 2 bf16/dword: 32 KB
    __shared__ __align__(16) unsigned int sXbf[4 * Kd * Cd / 2];
    // dot[k][c] fp32: 16 KB
    __shared__ float sDot[Kd * Cd];

    const int bn   = blockIdx.x;          // (b*N + n)
    const int tid  = threadIdx.x;
    const int lane = tid & 31;
    const int wave = tid >> 5;
    const int n16  = lane & 15;           // row (A) / col (B,C/D) within tile
    const int h    = lane >> 4;           // lane half

    // -------- Phase 1: global fp32 -> LDS bf16 (coalesced) --------
    const float2* x0p = (const float2*)(x0 + (size_t)bn * Kd * Cd);
    const float2* x1p = (const float2*)(x1 + (size_t)bn * Kd * 3 * Cd);
    for (int i = tid; i < 8192; i += 256) {
        int d, k, cc2;  // cc2 = c/2
        float2 v;
        if (i < 2048) {                      // x0 block: [32][128]
            k = i >> 6; cc2 = i & 63; d = 0;
            v = x0p[k * 64 + cc2];
        } else {                             // x1 block: [32][3][128]
            int ii = i - 2048;
            k = ii / 192; int rem = ii % 192;
            int j = rem >> 6; cc2 = rem & 63;
            d = 1 + j;
            v = x1p[(k * 3 + j) * 64 + cc2];
        }
        sXbf[(d * Kd + k) * (Cd / 2) + cc2] = f2bf(v.x) | (f2bf(v.y) << 16);
    }
    __syncthreads();

    // -------- Phase 2: WMMA projection fused with dot accumulation --------
    const uint4* sA  = (const uint4*)sXbf;
    const uint4* wtq = (const uint4*)wt;
    const float* x0b = x0 + (size_t)bn * Kd * Cd;
    const float* x1b = x1 + (size_t)bn * Kd * 3 * Cd;

    #pragma unroll
    for (int pp = 0; pp < 2; ++pp) {
        const int p     = wave * 2 + pp;   // 16 (mtile,ntile) jobs, 2 per wave
        const int mtile = p >> 3;          // 0..1  (k rows 0-15 / 16-31)
        const int ntile = p & 7;           // 0..7  (c cols)
        const int ncol  = ntile * 16 + n16;

        V8F dotAcc;
        #pragma unroll
        for (int r = 0; r < 8; ++r) dotAcc.f[r] = 0.0f;

        #pragma unroll
        for (int d = 0; d < 4; ++d) {
            V8F acc;
            const float bias = (d == 0) ? b0[ncol] : 0.0f;
            #pragma unroll
            for (int r = 0; r < 8; ++r) acc.f[r] = bias;

            const uint4* wbase = wtq + (d == 0 ? 0 : (Cd * Cd / 8));

            #pragma unroll
            for (int ks = 0; ks < 4; ++ks) {   // K=128 in 4 steps of 32
                V16B a, b;
                // A: 16-bit A-matrix layout — lane m=n16, K groups of 8
                // split by lane half: [8h, 8h+8) then [16+8h, 16+8h+8)
                int aIdx = ((d * Kd + mtile * 16 + n16) * Cd
                            + ks * 32 + 8 * h) >> 3;
                a.q[0] = sA[aIdx];
                a.q[1] = sA[aIdx + 2];
                // B: from W^T — lane col=ncol holds K=[8h, 8h+8) then [16+8h, 16+8h+8)
                int bIdx = (ncol * Cd + ks * 32 + 8 * h) >> 3;
                b.q[0] = wbase[bIdx];
                b.q[1] = wbase[bIdx + 2];
                acc.v = __builtin_amdgcn_wmma_f32_16x16x32_bf16(
                    false, a.v, false, b.v, (short)0, acc.v, false, false);
            }

            // dot[k,c] += x[k,d,c] * proj[d][k,c] straight from accumulator
            #pragma unroll
            for (int r = 0; r < 8; ++r) {
                int k = mtile * 16 + r + 8 * h;          // C/D layout row
                float xv = (d == 0)
                    ? x0b[k * Cd + ncol]
                    : x1b[(k * 3 + (d - 1)) * Cd + ncol];
                dotAcc.f[r] += xv * acc.f[r];
            }
        }

        #pragma unroll
        for (int r = 0; r < 8; ++r) {
            int k = mtile * 16 + r + 8 * h;
            sDot[k * Cd + ncol] = dotAcc.f[r];
        }
    }
    __syncthreads();

    // -------- Phase 3: per-channel argmax over K, gather x --------
    if (tid < Cd) {
        const int c = tid;
        float best = sDot[c];
        int   kb   = 0;
        for (int k = 1; k < Kd; ++k) {
            float v = sDot[k * Cd + c];
            if (v > best) { best = v; kb = k; }   // first-max wins, like argmax
        }
        out0[(size_t)bn * Cd + c] = x0b[kb * Cd + c];
        #pragma unroll
        for (int j = 0; j < 3; ++j)
            out1[((size_t)bn * 3 + j) * Cd + c] = x1b[(kb * 3 + j) * Cd + c];
    }
}

// ---------------------------------------------------------------------------
extern "C" void kernel_launch(void* const* d_in, const int* in_sizes, int n_in,
                              void* d_out, int out_size, void* d_ws, size_t ws_size,
                              hipStream_t stream) {
    const float* x0 = (const float*)d_in[0];   // [B,N,K,1,C]
    const float* x1 = (const float*)d_in[1];   // [B,N,K,3,C]
    const float* W0 = (const float*)d_in[2];   // [C,C]
    const float* b0 = (const float*)d_in[3];   // [C]
    const float* W1 = (const float*)d_in[4];   // [C,C]

    unsigned short* wt = (unsigned short*)d_ws;        // 64 KB: W0^T | W1^T bf16
    float* out0 = (float*)d_out;                       // y0: [B,N,1,C]
    float* out1 = out0 + (size_t)Bd * Nd * Cd;         // y1: [B,N,3,C]

    prep_WT<<<32, 256, 0, stream>>>(W0, W1, wt);
    vn_maxpool_kernel<<<Bd * Nd, 256, 0, stream>>>(x0, x1, b0, wt, out0, out1);
}
